// decoder_42090679501122
// MI455X (gfx1250) — compile-verified, matrix operand order checked by reading, not run on hardware
//
#include <hip/hip_runtime.h>
#include <stdint.h>

// CDNA5 (gfx1250) fused decoder: bf16 WMMA, wave32, LDS-staged weights,
// async global->LDS staging where the toolchain supports it.
typedef __attribute__((ext_vector_type(16))) __bf16 v16bf;
typedef __attribute__((ext_vector_type(8)))  float  v8f;
typedef int v4i __attribute__((vector_size(16)));   // matches async-lds builtin param

#define N_COARSE 12500
#define N_FINE   50000

// dynamic LDS: 16384 uint4 (256 KB) packed W_cur|W_last + equ[16][264] f32
#define SMEM_UINT4 16384
#define SMEM_BYTES (SMEM_UINT4 * 16 + 16 * 264 * 4)

union V16 {
  uint32_t u[8];
  uint4    q[2];
  v16bf    v;
};

// f32 -> bf16 (RNE), two values packed into one u32
#if __has_builtin(__builtin_amdgcn_cvt_pk_bf16_f32)
__device__ __forceinline__ uint32_t pk_bf16(float lo, float hi) {
  auto r = __builtin_amdgcn_cvt_pk_bf16_f32(lo, hi);
  return __builtin_bit_cast(uint32_t, r);
}
#else
__device__ __forceinline__ uint32_t pk_bf16(float lo, float hi) {
  uint32_t a = __float_as_uint(lo);
  uint32_t b = __float_as_uint(hi);
  a = (a + 0x7FFFu + ((a >> 16) & 1u)) >> 16;
  b = (b + 0x7FFFu + ((b >> 16) & 1u)) >> 16;
  return (a & 0xFFFFu) | (b << 16);
}
#endif

__device__ __forceinline__ void pack_v16(V16& d, float4 a0, float4 a1, float4 a2, float4 a3) {
  d.u[0] = pk_bf16(a0.x, a0.y);
  d.u[1] = pk_bf16(a0.z, a0.w);
  d.u[2] = pk_bf16(a1.x, a1.y);
  d.u[3] = pk_bf16(a1.z, a1.w);
  d.u[4] = pk_bf16(a2.x, a2.y);
  d.u[5] = pk_bf16(a2.z, a2.w);
  d.u[6] = pk_bf16(a3.x, a3.y);
  d.u[7] = pk_bf16(a3.z, a3.w);
}

// ---------------------------------------------------------------------------
// Kernel 0: pack W_cur_equ, W_last_equ (256x256) and W_mlp (768x512) f32 ->
// bf16 in WMMA B-operand lane layout:
//   packed[((kt*NT + nt)*32 + lane)*16 + e]  =  W[c, d]
//   lanes 0-15 : N = lane,    K = kt*32 + {0..7 | 16..23}
//   lanes 16-31: N = lane-16, K = kt*32 + {8..15 | 24..31}
// ws layout (uint16): [0,65536)=W_cur  [65536,131072)=W_last  [131072,524288)=W_mlp
// ---------------------------------------------------------------------------
__global__ void pack_weights(const float* __restrict__ Wc,
                             const float* __restrict__ Wl,
                             const float* __restrict__ Wm,
                             uint16_t* __restrict__ out) {
  int t = blockIdx.x * blockDim.x + threadIdx.x;
  const float* src;
  int idx, NT, ncols;
  if (t < 65536)        { src = Wc; idx = t;          NT = 16; ncols = 256; }
  else if (t < 131072)  { src = Wl; idx = t - 65536;  NT = 16; ncols = 256; }
  else                  { src = Wm; idx = t - 131072; NT = 32; ncols = 512; }
  int e    = idx & 15;
  int lane = (idx >> 4) & 31;
  int tile = idx >> 9;                 // kt*NT + nt
  int nt   = tile % NT;
  int kt   = tile / NT;
  int hi   = (lane >> 4) & 1;
  int c = kt * 32 + ((e < 8) ? e : 16 + (e - 8)) + hi * 8;
  int d = nt * 16 + (lane & 15);
  uint32_t a = __float_as_uint(src[(size_t)c * ncols + d]);
  a = (a + 0x7FFFu + ((a >> 16) & 1u)) >> 16;
  out[t] = (uint16_t)a;
}

// ---------------------------------------------------------------------------
// Kernel 1: fused equ-bilinear + MLP GEMM. Block = 16 fine points, 256 thr.
// ---------------------------------------------------------------------------
__global__ void __launch_bounds__(256) fused_decoder(
    const float* __restrict__ last_inv,    // [12500, 256]
    const float* __restrict__ cur_inv,     // [50000, 256]
    const float* __restrict__ last_equ,    // [12500, 16, 256]
    const float* __restrict__ cur_equ,     // [50000, 16, 256]
    const int*   __restrict__ upsampling,  // [50000]
    const uint4* __restrict__ pWpair,      // packed bf16 W_cur | W_last (contiguous)
    const uint4* __restrict__ pWm,         // packed bf16 W_mlp
    float* __restrict__ out)               // [50000, 512]
{
  extern __shared__ uint4 smem[];
  const uint4* wcl     = smem;                        // [0,8192)=W_cur  [8192,16384)=W_last
  float*       lds_equ = (float*)(smem + SMEM_UINT4); // equ[16][264] (stride 264: no bank conflicts)

  const int lane  = threadIdx.x & 31;
  const int wave  = threadIdx.x >> 5;
  const int row   = lane & 15;             // M (and N) index within half-wave
  const int hi8   = (lane >> 4) * 8;       // K sub-offset for upper half-wave
  const int mbase = blockIdx.x * 16;

  // ---- Stage 0: global -> LDS staging of the 256 KB packed weight pair ----
#if __has_builtin(__builtin_amdgcn_global_load_async_to_lds_b128)
  {
    v4i* g = (v4i*)pWpair;   // generic pointers; builtin takes int4*
    v4i* l = (v4i*)smem;
#pragma unroll 4
    for (int i = threadIdx.x; i < SMEM_UINT4; i += 256) {
      __builtin_amdgcn_global_load_async_to_lds_b128(g + i, l + i, 0, 0);
    }
#if __has_builtin(__builtin_amdgcn_s_wait_asynccnt)
    __builtin_amdgcn_s_wait_asynccnt(0);
#else
    asm volatile("s_wait_asynccnt 0x0" ::: "memory");
#endif
  }
#else
#pragma unroll 4
  for (int i = threadIdx.x; i < SMEM_UINT4; i += 256) smem[i] = pWpair[i];
#endif
  __syncthreads();

  // ================= Stage 1: equ[p][256] for the block's 16 points =======
  {
    // warm L2->L1 for the gathered second point while the first computes
    const int n1 = mbase + wave * 2 + 1;
    const int u1 = upsampling[n1];
    __builtin_prefetch(last_equ + ((size_t)u1 * 16 + row) * 256, 0, 3);
    __builtin_prefetch(cur_equ  + ((size_t)n1 * 16 + row) * 256, 0, 3);
  }

  for (int pp = 0; pp < 2; ++pp) {
    const int p = wave * 2 + pp;
    const int n = mbase + p;
    const int u = upsampling[n];

    const float* ac = cur_equ  + ((size_t)n * 16 + row) * 256;
    const float* al = last_equ + ((size_t)u * 16 + row) * 256;

    // Preload + convert A operands for all 8 K-steps (K = 256 = 8 x 32)
    V16 Acur[8], Alast[8];
#pragma unroll
    for (int kt = 0; kt < 8; ++kt) {
      const int c1 = kt * 32 + hi8;
      const int c2 = c1 + 16;
      pack_v16(Acur[kt],
               *(const float4*)(ac + c1), *(const float4*)(ac + c1 + 4),
               *(const float4*)(ac + c2), *(const float4*)(ac + c2 + 4));
      pack_v16(Alast[kt],
               *(const float4*)(al + c1), *(const float4*)(al + c1 + 4),
               *(const float4*)(al + c2), *(const float4*)(al + c2 + 4));
    }

#pragma unroll 1
    for (int nt = 0; nt < 16; ++nt) {
      v8f acc_c = {};
      v8f acc_l = {};
#pragma unroll
      for (int kt = 0; kt < 8; ++kt) {
        V16 Bc, Bl;
        const int bidx = ((kt * 16 + nt) * 32 + lane) * 2;
        Bc.q[0] = wcl[bidx];        Bc.q[1] = wcl[bidx + 1];
        Bl.q[0] = wcl[8192 + bidx]; Bl.q[1] = wcl[8192 + bidx + 1];
        acc_c = __builtin_amdgcn_wmma_f32_16x16x32_bf16(
                    false, Acur[kt].v,  false, Bc.v, (short)0, acc_c, false, false);
        acc_l = __builtin_amdgcn_wmma_f32_16x16x32_bf16(
                    false, Alast[kt].v, false, Bl.v, (short)0, acc_l, false, false);
      }
      // equ = mean over M(=basis) of ce*le.  D layout: lane holds N=row,
      // M = v (+8 for upper half-wave) -> per-lane sum + swap-16 combine.
      float s = 0.f;
#pragma unroll
      for (int v = 0; v < 8; ++v) s += acc_c[v] * acc_l[v];
      s += __shfl_xor(s, 16, 32);
      if (lane < 16) lds_equ[p * 264 + nt * 16 + lane] = s * 0.0625f;
    }
  }
  __syncthreads();

  // ================= Stage 2: out[m,:] = [li | cur_inv | equ] @ W_mlp =====
  const int m  = mbase + row;
  const int u2 = upsampling[m];
  const float* gli = last_inv + (size_t)u2 * 256;
  const float* gci = cur_inv  + (size_t)m  * 256;
  __builtin_prefetch(gli, 0, 3);

  v8f acc[4] = {v8f{}, v8f{}, v8f{}, v8f{}};
#pragma unroll 1
  for (int kt = 0; kt < 24; ++kt) {
    V16 A;
    if (kt < 16) {
      const float* s = (kt < 8) ? gli : gci;
      const int c1 = (kt & 7) * 32 + hi8;
      const int c2 = c1 + 16;
      pack_v16(A,
               *(const float4*)(s + c1), *(const float4*)(s + c1 + 4),
               *(const float4*)(s + c2), *(const float4*)(s + c2 + 4));
    } else {
      const float* s = lds_equ + row * 264;
      const int c1 = (kt - 16) * 32 + hi8;
      const int c2 = c1 + 16;
      pack_v16(A,
               *(const float4*)(s + c1), *(const float4*)(s + c1 + 4),
               *(const float4*)(s + c2), *(const float4*)(s + c2 + 4));
    }
#pragma unroll
    for (int j = 0; j < 4; ++j) {
      const int nt = wave + 8 * j;                 // 8 waves x 4 tiles = 32 N-tiles (512 cols)
      V16 B;
      const int bidx = ((kt * 32 + nt) * 32 + lane) * 2;
      B.q[0] = pWm[bidx]; B.q[1] = pWm[bidx + 1];
      acc[j] = __builtin_amdgcn_wmma_f32_16x16x32_bf16(
                   false, A.v, false, B.v, (short)0, acc[j], false, false);
    }
  }

  // Store: D layout -> lane owns column nt*16+row, rows mbase + v (+8 upper half)
  const int mrow = mbase + ((lane >> 4) ? 8 : 0);
#pragma unroll
  for (int j = 0; j < 4; ++j) {
    const int col = (wave + 8 * j) * 16 + row;
#pragma unroll
    for (int v = 0; v < 8; ++v) {
      out[(size_t)(mrow + v) * 512 + col] = acc[j][v];
    }
  }
}

// ---------------------------------------------------------------------------
extern "C" void kernel_launch(void* const* d_in, const int* in_sizes, int n_in,
                              void* d_out, int out_size, void* d_ws, size_t ws_size,
                              hipStream_t stream) {
  const float* last_inv   = (const float*)d_in[0];
  const float* cur_inv    = (const float*)d_in[1];
  const float* last_equ   = (const float*)d_in[2];
  const float* cur_equ    = (const float*)d_in[3];
  const int*   upsampling = (const int*)  d_in[4];
  const float* W_last_equ = (const float*)d_in[5];
  const float* W_cur_equ  = (const float*)d_in[6];
  const float* W_mlp      = (const float*)d_in[7];

  uint16_t* ws  = (uint16_t*)d_ws;
  uint16_t* pWm = ws + 131072;    // after the 2x65536 pair

  // allow >64KB dynamic LDS (WGP has 320 KB); deterministic, capture-safe
  (void)hipFuncSetAttribute((const void*)fused_decoder,
                            hipFuncAttributeMaxDynamicSharedMemorySize, SMEM_BYTES);

  // 524288 packed elements total = 2048 blocks x 256 threads
  pack_weights<<<2048, 256, 0, stream>>>(W_cur_equ, W_last_equ, W_mlp, ws);

  fused_decoder<<<N_FINE / 16, 256, SMEM_BYTES, stream>>>(
      last_inv, cur_inv, last_equ, cur_equ, upsampling,
      (const uint4*)ws, (const uint4*)pWm,
      (float*)d_out);
}
